// Head_90933047591318
// MI455X (gfx1250) — matrix operand order repaired and checked by reading, not compiled, not run on hardware
//
#include <hip/hip_runtime.h>

// ---------------------------------------------------------------------------
// Causal single-head attention, flash-attention style, CDNA5 (gfx1250).
//   B=8, T=4096, C=512, H=64.  fp32 in/out, f16 WMMA with f32 accumulate.
// Pass 0: W{q,k,v} -> f16 transposed [H,C]
// Pass 1: q,k = x@W ([T,H] f16), v = x@Wv stored TRANSPOSED ([H,T] f16) so
//         attention V B-frags are contiguous.  W^T slices staged by TDM.
// Pass 2: flash attention; K ([32keys,64h]) and V^T ([64h,32keys]) chunks
//         staged to LDS by TDM, double-buffered (DMA overlaps compute).
// ---------------------------------------------------------------------------

typedef __attribute__((ext_vector_type(16))) _Float16 v16h;
typedef __attribute__((ext_vector_type(8)))  _Float16 v8h;
typedef __attribute__((ext_vector_type(8)))  float    v8f;
typedef __attribute__((ext_vector_type(4)))  unsigned int v4u;
typedef __attribute__((ext_vector_type(8)))  int      v8i;
typedef __attribute__((ext_vector_type(4)))  int      v4i;

#define BDIM 8
#define TDIM 4096
#define CDIM 512
#define HDIM 64

__device__ __forceinline__ v8f wmma16x16x32(v16h a, v16h b, v8f c) {
  return __builtin_amdgcn_wmma_f32_16x16x32_f16(
      false, a, false, b, (short)0, c, false, false);
}

// ---------------------------------------------------------------------------
// Tensor Data Mover: 2D f16 tile (tile_w x tile_h elements) from a row-major
// tensor (row length = tensor_w elements) into LDS, rows packed contiguously.
// D# per CDNA5 ISA ch.8.  6-arg builtin form on this toolchain.
// ---------------------------------------------------------------------------
#if __has_builtin(__builtin_amdgcn_tensor_load_to_lds)
#define HAVE_TDM 1
__device__ __forceinline__ void tdm_load_2d(unsigned lds_off,
                                            const _Float16* gsrc,
                                            unsigned tile_w, unsigned tile_h,
                                            unsigned tensor_w,
                                            unsigned tensor_h) {
  const unsigned long long ga = (unsigned long long)(uintptr_t)gsrc;
  v4u g0;
  g0[0] = 1u;                                       // count=1, user descriptor
  g0[1] = lds_off;                                  // lds_addr (bytes)
  g0[2] = (unsigned)(ga & 0xFFFFFFFFu);             // global_addr[31:0]
  g0[3] = (unsigned)((ga >> 32) & 0x01FFFFFFu)      // global_addr[56:32]
          | (2u << 30);                             // type=2 (image)
  v8i g1;
  g1[0] = (int)(1u << 16);                          // data_size=1 -> 2 bytes
  g1[1] = (int)((tensor_w & 0xFFFFu) << 16);        // tensor_dim0 lo16
  g1[2] = (int)((tensor_w >> 16) | ((tensor_h & 0xFFFFu) << 16));
  g1[3] = (int)(((tensor_h >> 16) & 0xFFFFu) | (tile_w << 16)); // tile_dim0
  g1[4] = (int)tile_h;                              // tile_dim1, tile_dim2=0
  g1[5] = (int)tensor_w;                            // tensor_dim0_stride lo32
  g1[6] = 0;
  g1[7] = 0;
  const v4i z4 = {0, 0, 0, 0};
  const v8i z8 = {0, 0, 0, 0, 0, 0, 0, 0};
  __builtin_amdgcn_tensor_load_to_lds(g0, g1, z4, z4, z8, 0);
}
#else
#define HAVE_TDM 0
#endif

// ---------------------------------------------------------------------------
// Pass 0: transpose + convert the three weights to f16 [H][C].
// ---------------------------------------------------------------------------
__global__ __launch_bounds__(256) void wt_kernel(
    const float* __restrict__ Wq, const float* __restrict__ Wk,
    const float* __restrict__ Wv,
    _Float16* __restrict__ WqT, _Float16* __restrict__ WkT,
    _Float16* __restrict__ WvT)
{
  const int i = blockIdx.x * 256 + threadIdx.x;   // 3*C*H = 98304 threads
  const int w = i / (CDIM * HDIM);
  const int r = i % (CDIM * HDIM);
  const int n = r / CDIM;
  const int c = r % CDIM;
  const float* W = (w == 0) ? Wq : (w == 1) ? Wk : Wv;
  _Float16*  WT = (w == 0) ? WqT : (w == 1) ? WkT : WvT;
  WT[(size_t)n * CDIM + c] = (_Float16)W[(size_t)c * HDIM + n];
}

// ---------------------------------------------------------------------------
// Pass 1: QKV projection.  Block = 8 waves = 128 rows of x; all waves share
// the per-kc W^T slice (3 x 64x32 f16 = 12 KB) staged to LDS by TDM, double-
// buffered.  One wave -> 16x64 tile of q, k and v.
// q,k stored [B,T,H]; v stored transposed [B,H,T].
// ---------------------------------------------------------------------------
__global__ __launch_bounds__(256) void qkv_proj_kernel(
    const float* __restrict__ x,
    const _Float16* __restrict__ WqT, const _Float16* __restrict__ WkT,
    const _Float16* __restrict__ WvT,
    _Float16* __restrict__ qo, _Float16* __restrict__ ko,
    _Float16* __restrict__ vT)
{
  __shared__ _Float16 wtile[2][3][64 * 32];   // [buf][weight][row*32+c]

  const int lane = threadIdx.x & 31;
  const int wave = threadIdx.x >> 5;
  const int tile = blockIdx.x * 8 + wave;     // 2048 16-row tiles
  const int row0 = tile << 4;
  const int m = lane & 15;
  const int h = lane >> 4;

  v8f cq[4] = {}, ck[4] = {}, cv[4] = {};

  const float* xrow = x + (size_t)(row0 + m) * CDIM;

#if HAVE_TDM
  if (wave == 0) {
    tdm_load_2d((unsigned)(uintptr_t)&wtile[0][0][0], WqT, 32, 64, CDIM, HDIM);
    tdm_load_2d((unsigned)(uintptr_t)&wtile[0][1][0], WkT, 32, 64, CDIM, HDIM);
    tdm_load_2d((unsigned)(uintptr_t)&wtile[0][2][0], WvT, 32, 64, CDIM, HDIM);
  }
#endif
  int cur = 0;

  for (int kc = 0; kc < CDIM; kc += 32) {
#if HAVE_TDM
    if (wave == 0) __builtin_amdgcn_s_wait_tensorcnt(0);
    __syncthreads();
    if (wave == 0 && kc + 32 < CDIM) {
      const int nb = cur ^ 1;
      tdm_load_2d((unsigned)(uintptr_t)&wtile[nb][0][0], WqT + kc + 32,
                  32, 64, CDIM, HDIM);
      tdm_load_2d((unsigned)(uintptr_t)&wtile[nb][1][0], WkT + kc + 32,
                  32, 64, CDIM, HDIM);
      tdm_load_2d((unsigned)(uintptr_t)&wtile[nb][2][0], WvT + kc + 32,
                  32, 64, CDIM, HDIM);
    }
#else
    {
      const int t = threadIdx.x;              // 256 thr: 64 rows x 4 parts
      const int row = t >> 2, part = (t & 3) << 3;
      *(v8h*)&wtile[0][0][row * 32 + part] =
          *(const v8h*)(WqT + (size_t)row * CDIM + kc + part);
      *(v8h*)&wtile[0][1][row * 32 + part] =
          *(const v8h*)(WkT + (size_t)row * CDIM + kc + part);
      *(v8h*)&wtile[0][2][row * 32 + part] =
          *(const v8h*)(WvT + (size_t)row * CDIM + kc + part);
    }
    __syncthreads();
#endif

    // A fragment: two 32-byte contiguous fp32 runs of x, convert to f16
    const v8f xa = *(const v8f*)(xrow + kc + (h << 4));
    const v8f xb = *(const v8f*)(xrow + kc + 8 + (h << 4));
    v16h a;
#pragma unroll
    for (int i = 0; i < 8; ++i) {
      a[i]     = (_Float16)xa[i];
      a[8 + i] = (_Float16)xb[i];
    }

#pragma unroll
    for (int j = 0; j < 4; ++j) {             // four 16-col tiles of H
      const int wrow = ((j << 4) + m) * 32 + (h << 4);
      const v16h bq = *(const v16h*)&wtile[cur][0][wrow];
      const v16h bk = *(const v16h*)&wtile[cur][1][wrow];
      const v16h bv = *(const v16h*)&wtile[cur][2][wrow];
      cq[j] = wmma16x16x32(a, bq, cq[j]);
      ck[j] = wmma16x16x32(a, bk, ck[j]);
      cv[j] = wmma16x16x32(a, bv, cv[j]);
    }
    __syncthreads();
#if HAVE_TDM
    cur ^= 1;
#endif
  }

  // q,k: [B,T,H] row-major.  v: transposed [B,H,T].
  const int bb = row0 >> 12;                  // row0 / TDIM
  const int t0 = row0 & (TDIM - 1);
  _Float16* vTb = vT + (size_t)bb * HDIM * TDIM;
#pragma unroll
  for (int j = 0; j < 4; ++j)
#pragma unroll
    for (int r = 0; r < 8; ++r) {
      const size_t idx = (size_t)(row0 + r + (h << 3)) * HDIM + (j << 4) + m;
      qo[idx] = (_Float16)cq[j][r];
      ko[idx] = (_Float16)ck[j][r];
      vTb[(size_t)((j << 4) + m) * TDIM + t0 + r + (h << 3)] =
          (_Float16)cv[j][r];
    }
}

// ---------------------------------------------------------------------------
// Pass 2: flash attention.  Block = 4 waves = 64 query rows; K and V^T
// 32-key chunks staged to LDS by TDM, double-buffered.
// Per chunk & wave: 4 wmma S=QK^T, online softmax, P transposed via per-wave
// LDS, 4 wmma O += P V.  All fragment reads from LDS are contiguous b128s.
// ---------------------------------------------------------------------------
__global__ __launch_bounds__(128) void flash_attn_kernel(
    const _Float16* __restrict__ qg, const _Float16* __restrict__ kg,
    const _Float16* __restrict__ vT, float* __restrict__ out)
{
  __shared__ _Float16 ktile[2][32 * 64];    // rows = keys, cols = hdim
  __shared__ _Float16 vtile[2][64 * 32];    // rows = hdim, cols = keys (V^T)
  __shared__ _Float16 pshared[4][16][36];   // per-wave P tile (+pad)

  const int lane = threadIdx.x & 31;
  const int wave = threadIdx.x >> 5;
  const int b   = blockIdx.x >> 6;          // 64 blocks per batch
  const int qb0 = (blockIdx.x & 63) << 6;   // first query row of the block
  const int q0  = qb0 + (wave << 4);        // this wave's 16-query tile
  const int m = lane & 15;
  const int h = lane >> 4;

  const _Float16* qb  = qg + ((size_t)b * TDIM + q0) * HDIM;
  const _Float16* kb  = kg + (size_t)b * TDIM * HDIM;
  const _Float16* vtb = vT + (size_t)b * HDIM * TDIM;

  // Q as two A-fragments, two contiguous 16B runs each
  v16h qa[2];
#pragma unroll
  for (int f = 0; f < 2; ++f) {
    const v8h lo = *(const v8h*)(qb + m * HDIM + (f << 5) + (h << 4));
    const v8h hi = *(const v8h*)(qb + m * HDIM + (f << 5) + 8 + (h << 4));
#pragma unroll
    for (int i = 0; i < 8; ++i) { qa[f][i] = lo[i]; qa[f][8 + i] = hi[i]; }
  }

  float mrow[8], lrow[8];
  v8f o[4] = {};
#pragma unroll
  for (int r = 0; r < 8; ++r) { mrow[r] = -1e30f; lrow[r] = 0.0f; }

  const float scale = 0.044194173824159216f;   // C^-0.5
  const int kmax_w = q0 + 15;                  // last valid key for this wave
  const int kend   = qb0 + 63;                 // last key any wave needs

#if HAVE_TDM
  if (wave == 0) {
    tdm_load_2d((unsigned)(uintptr_t)&ktile[0][0], kb, 64, 32, 64, TDIM);
    tdm_load_2d((unsigned)(uintptr_t)&vtile[0][0], vtb, 32, 64, TDIM, HDIM);
  }
#endif
  int cur = 0;

  for (int c0 = 0; c0 <= kend; c0 += 32) {
#if HAVE_TDM
    if (wave == 0) __builtin_amdgcn_s_wait_tensorcnt(0);
    __syncthreads();
    if (wave == 0 && c0 + 32 <= kend) {       // prefetch next chunk
      const int nb = cur ^ 1;
      tdm_load_2d((unsigned)(uintptr_t)&ktile[nb][0],
                  kb + (size_t)(c0 + 32) * HDIM, 64, 32, 64, TDIM);
      tdm_load_2d((unsigned)(uintptr_t)&vtile[nb][0],
                  vtb + c0 + 32, 32, 64, TDIM, HDIM);
    }
#else
    {
      const int t = threadIdx.x;              // 128 threads
      *(v16h*)(&ktile[0][0] + t * 16) =
          *(const v16h*)(kb + (size_t)c0 * HDIM + t * 16);
      const int row = t >> 1, part = (t & 1) << 4;   // 64 rows x 2 parts
      *(v16h*)(&vtile[0][0] + row * 32 + part) =
          *(const v16h*)(vtb + (size_t)row * TDIM + c0 + part);
    }
    __syncthreads();
#endif

    if (c0 <= kmax_w) {                       // wave-uniform causal skip
      const _Float16* kt = &ktile[cur][0];
      const _Float16* vt = &vtile[cur][0];

      // ---- S = Q K^T, two 16x16 tiles ----
      v8f s0 = {}, s1 = {};
      {
        const _Float16* kl0 = kt + (size_t)m * HDIM;
        const v16h b0a = *(const v16h*)(kl0 + (h << 4));
        const v16h b0b = *(const v16h*)(kl0 + 32 + (h << 4));
        s0 = wmma16x16x32(qa[0], b0a, s0);
        s0 = wmma16x16x32(qa[1], b0b, s0);
        const _Float16* kl1 = kt + (size_t)(16 + m) * HDIM;
        const v16h b1a = *(const v16h*)(kl1 + (h << 4));
        const v16h b1b = *(const v16h*)(kl1 + 32 + (h << 4));
        s1 = wmma16x16x32(qa[0], b1a, s1);
        s1 = wmma16x16x32(qa[1], b1b, s1);
      }

      // ---- causal mask + online softmax (rows r+8h, cols m / 16+m) ----
#pragma unroll
      for (int r = 0; r < 8; ++r) {
        const int rowq = q0 + r + (h << 3);
        float x0 = (c0 + m      <= rowq) ? s0[r] * scale : -1e30f;
        float x1 = (c0 + 16 + m <= rowq) ? s1[r] * scale : -1e30f;
        float mx = fmaxf(x0, x1);
#pragma unroll
        for (int off = 1; off < 16; off <<= 1)
          mx = fmaxf(mx, __shfl_xor(mx, off, 32));
        const float mnew  = fmaxf(mrow[r], mx);
        const float p0    = __expf(x0 - mnew);
        const float p1    = __expf(x1 - mnew);
        const float alpha = __expf(mrow[r] - mnew);
        float ps = p0 + p1;
#pragma unroll
        for (int off = 1; off < 16; off <<= 1)
          ps += __shfl_xor(ps, off, 32);
        lrow[r] = lrow[r] * alpha + ps;
        mrow[r] = mnew;
#pragma unroll
        for (int j = 0; j < 4; ++j) o[j][r] *= alpha;
        pshared[wave][r + (h << 3)][m]      = (_Float16)p0;
        pshared[wave][r + (h << 3)][16 + m] = (_Float16)p1;
      }

      // same-wave DS ops execute in order; wait so all lanes' stores land
      asm volatile("s_wait_dscnt 0" ::: "memory");

      // ---- P back as A-fragment (16 q-rows x 32 keys) ----
      v16h pa;
#pragma unroll
      for (int t = 0; t < 8; ++t) {
        const int kk = (h << 4) + (t << 1);
        pa[2 * t]     = pshared[wave][m][kk];
        pa[2 * t + 1] = pshared[wave][m][kk + 1];
      }

      // ---- O += P V  (V^T tile: one contiguous v16h per lane per frag) ----
#pragma unroll
      for (int j = 0; j < 4; ++j) {
        const v16h bv =
            *(const v16h*)(vt + (size_t)((j << 4) + m) * 32 + (h << 4));
        o[j] = wmma16x16x32(pa, bv, o[j]);
      }
    }
    __syncthreads();   // all reads of buf `cur` done before it is re-staged
#if HAVE_TDM
    cur ^= 1;
#endif
  }

  // ---- normalize and store fp32 output ----
#pragma unroll
  for (int r = 0; r < 8; ++r) {
    const float inv = 1.0f / lrow[r];
    const size_t rowo = ((size_t)b * TDIM + q0 + r + (h << 3)) * HDIM;
#pragma unroll
    for (int j = 0; j < 4; ++j)
      out[rowo + (j << 4) + m] = o[j][r] * inv;
  }
}

// ---------------------------------------------------------------------------
extern "C" void kernel_launch(void* const* d_in, const int* in_sizes, int n_in,
                              void* d_out, int out_size, void* d_ws, size_t ws_size,
                              hipStream_t stream) {
  (void)in_sizes; (void)n_in; (void)out_size; (void)ws_size;
  const float* x  = (const float*)d_in[0];
  const float* Wq = (const float*)d_in[1];
  const float* Wk = (const float*)d_in[2];
  const float* Wv = (const float*)d_in[3];

  const size_t nqkv = (size_t)BDIM * TDIM * HDIM;   // 2M elements each
  const size_t nw   = (size_t)CDIM * HDIM;          // 32K elements each
  _Float16* qw  = (_Float16*)d_ws;                  // f16 q,k: [B,T,H]
  _Float16* kw  = qw + nqkv;
  _Float16* vw  = kw + nqkv;                        // f16 v: [B,H,T]
  _Float16* wqt = vw + nqkv;                        // f16 W^T: 3 x 64 KB
  _Float16* wkt = wqt + nw;
  _Float16* wvt = wkt + nw;

  wt_kernel<<<384, 256, 0, stream>>>(Wq, Wk, Wv, wqt, wkt, wvt);
  qkv_proj_kernel<<<256, 256, 0, stream>>>(x, wqt, wkt, wvt, qw, kw, vw);
  flash_attn_kernel<<<512, 128, 0, stream>>>(qw, kw, vw, (float*)d_out);
}
